// SparseGaussian3DRefinementModule_58858231824545
// MI455X (gfx1250) — matrix operand order repaired and checked by reading, not compiled, not run on hardware
//
#include <hip/hip_runtime.h>
#include <math.h>

// ---------------------------------------------------------------------------
// Types for CDNA5 WMMA
// ---------------------------------------------------------------------------
typedef __attribute__((ext_vector_type(16))) __bf16     v16bf;
typedef __attribute__((ext_vector_type(8)))  float      v8f;
typedef __attribute__((ext_vector_type(4)))  unsigned   u32x4;

union U16 { v16bf v; u32x4 q[2]; };

__device__ __forceinline__ __bf16 f2bf(float f) {
    unsigned u = __float_as_uint(f);
    unsigned r = (u + 0x7FFFu + ((u >> 16) & 1u)) >> 16;   // round-to-nearest-even
    unsigned short s = (unsigned short)r;
    __bf16 out;
    __builtin_memcpy(&out, &s, 2);
    return out;
}

__device__ __forceinline__ float safe_sigmoid(float x) {
    x = fminf(fmaxf(x, -9.21f), 9.21f);
    return 1.0f / (1.0f + __expf(-x));
}

// ---------------------------------------------------------------------------
// Problem constants
// ---------------------------------------------------------------------------
#define NROWS   262144
#define GCOUNT  1048576L
#define DDIM    128
#define OUTDIM  86
#define ROWS_PER_BLOCK 128          // 8 waves x 16 rows
#define NBLOCKS (NROWS / ROWS_PER_BLOCK)   // 2048

// ---------------------------------------------------------------------------
// Dynamic LDS layout (bytes)
// ---------------------------------------------------------------------------
constexpr int W_OFF    = 0;
constexpr int W_BYTES  = 4 * 128 * 128 * 2;        // w1..w4 bf16, transposed [n][k]
constexpr int W5_OFF   = W_OFF + W_BYTES;          // 131072
constexpr int W5_BYTES = 16 * 128 * 2;             // first 16 cols of w5, transposed
constexpr int BV_OFF   = W5_OFF + W5_BYTES;        // 135168 : b1..b4 f32
constexpr int BV_BYTES = 4 * 128 * 4;
constexpr int B5_OFF   = BV_OFF + BV_BYTES;        // 137216 : b5[0..15], scale[0..15]
constexpr int SC_OFF   = B5_OFF + 64;
constexpr int LN_OFF   = B5_OFF + 128;             // 137344 : ln1_g, ln1_b, ln2_g, ln2_b
constexpr int LN_BYTES = 4 * 128 * 4;
constexpr int ACT_OFF  = LN_OFF + LN_BYTES;        // 139392 : 8 waves x 16x128 bf16
constexpr int ACT_BYTES = 8 * 16 * 128 * 2;
constexpr int OUTB_OFF = ACT_OFF + ACT_BYTES;      // 172160 : 8 waves x 16x16 f32
constexpr int OUTB_BYTES = 8 * 16 * 16 * 4;
constexpr int SRED_OFF = OUTB_OFF + OUTB_BYTES;    // 180352
constexpr int SMEM_TOTAL = SRED_OFF + 32;          // 180384 bytes (< 320KB WGP LDS)

// ---------------------------------------------------------------------------
// Fragment loaders (layouts per CDNA5 ISA §7.12.2)
// A 16x32 bf16 : lane l -> row m=l&15; half h=l>>4 ; elems 0..7 = K kb..kb+7,
//                elems 8..15 = K kb+16..kb+23 with kb = 32*kt + 8*h
// B 32x16 bf16 : lane l -> col n=(l&15)+16t ; same K order as A (kb = 32*kt + 8*h)
//                (weights stored transposed in LDS: wt[n*128 + k])
// ---------------------------------------------------------------------------
__device__ __forceinline__ void load_a(const __bf16* act, int lane, v16bf a[4]) {
    const int m = lane & 15, h = lane >> 4;
    const char* base = (const char*)(act + m * 128);
#pragma unroll
    for (int kt = 0; kt < 4; ++kt) {
        const int kb = kt * 32 + h * 8;
        U16 u;
        u.q[0] = *(const u32x4*)(base + kb * 2);
        u.q[1] = *(const u32x4*)(base + (kb + 16) * 2);
        a[kt] = u.v;
    }
}

__device__ __forceinline__ v16bf load_b(const __bf16* wt, int lane, int t, int kt) {
    const int n = (lane & 15) + 16 * t, h = lane >> 4;
    const char* p = (const char*)(wt + n * 128 + kt * 32 + h * 8);
    U16 u;
    u.q[0] = *(const u32x4*)p;
    u.q[1] = *(const u32x4*)(p + 32);
    return u.v;
}

// One 16x128 layer for this wave: act(in LDS, bf16) @ W + b, ReLU, optional LN.
__device__ __forceinline__ void mlp_layer(const __bf16* actin, __bf16* actout,
                                          const __bf16* wt, const float* bias,
                                          int lane, bool doLN,
                                          const float* lng, const float* lnb) {
    v16bf a[4];
    load_a(actin, lane, a);
    const int nlo = lane & 15, h = lane >> 4;
    v8f c[8];
#pragma unroll
    for (int t = 0; t < 8; ++t) {
        const float bn = bias[nlo + 16 * t];
        v8f acc;
#pragma unroll
        for (int r = 0; r < 8; ++r) acc[r] = bn;
#pragma unroll
        for (int kt = 0; kt < 4; ++kt) {
            v16bf b = load_b(wt, lane, t, kt);
            acc = __builtin_amdgcn_wmma_f32_16x16x32_bf16(
                false, a[kt], false, b, (short)0, acc, false, false);
        }
#pragma unroll
        for (int r = 0; r < 8; ++r) acc[r] = fmaxf(acc[r], 0.0f);
        c[t] = acc;
    }
    if (doLN) {
        // row m = r + 8*h lives on the 16 lanes of this half; reduce over
        // 8 in-register col-tiles then xor-shuffle across lanes 1,2,4,8.
#pragma unroll
        for (int r = 0; r < 8; ++r) {
            float s = 0.0f, q = 0.0f;
#pragma unroll
            for (int t = 0; t < 8; ++t) { const float v = c[t][r]; s += v; q += v * v; }
#pragma unroll
            for (int msk = 1; msk <= 8; msk <<= 1) {
                s += __shfl_xor(s, msk);
                q += __shfl_xor(q, msk);
            }
            const float mu  = s * (1.0f / 128.0f);
            const float var = q * (1.0f / 128.0f) - mu * mu;
            const float rs  = rsqrtf(var + 1e-5f);
#pragma unroll
            for (int t = 0; t < 8; ++t) {
                const int n = nlo + 16 * t;
                c[t][r] = (c[t][r] - mu) * rs * lng[n] + lnb[n];
            }
        }
    }
#pragma unroll
    for (int t = 0; t < 8; ++t)
#pragma unroll
        for (int r = 0; r < 8; ++r) {
            const int m = r + 8 * h, n = nlo + 16 * t;
            actout[m * 128 + n] = f2bf(c[t][r]);
        }
}

// ---------------------------------------------------------------------------
// Kernel 1: copy means/opacities into the output buffer
// ---------------------------------------------------------------------------
__global__ void copy_kernel(const float* __restrict__ means,
                            const float* __restrict__ opac,
                            float* __restrict__ out) {
    const long stride = (long)gridDim.x * blockDim.x;
    for (long i = (long)blockIdx.x * blockDim.x + threadIdx.x; i < GCOUNT * 3; i += stride)
        out[i] = means[i];
    for (long j = (long)blockIdx.x * blockDim.x + threadIdx.x; j < GCOUNT; j += stride)
        out[GCOUNT * 3 + j] = opac[j];
}

// ---------------------------------------------------------------------------
// Kernel 2: the fused MLP. 2048 blocks x 256 threads; each wave owns 16 rows.
// ---------------------------------------------------------------------------
__global__ void __launch_bounds__(256)
mlp_kernel(const float* __restrict__ inst, const float* __restrict__ anchor,
           const float* __restrict__ emb,
           const float* __restrict__ w1, const float* __restrict__ b1,
           const float* __restrict__ w2, const float* __restrict__ b2,
           const float* __restrict__ ln1g, const float* __restrict__ ln1b,
           const float* __restrict__ w3, const float* __restrict__ b3,
           const float* __restrict__ w4, const float* __restrict__ b4,
           const float* __restrict__ ln2g, const float* __restrict__ ln2b,
           const float* __restrict__ w5, const float* __restrict__ b5,
           const float* __restrict__ scale,
           float* __restrict__ ws_xyz, float* __restrict__ ws_opa,
           float* __restrict__ ws_part) {
    extern __shared__ char smem[];
    __bf16* Wt   = (__bf16*)(smem + W_OFF);
    __bf16* W5t  = (__bf16*)(smem + W5_OFF);
    float*  Bv   = (float*)(smem + BV_OFF);
    float*  B5s  = (float*)(smem + B5_OFF);
    float*  SCs  = (float*)(smem + SC_OFF);
    float*  LNp  = (float*)(smem + LN_OFF);
    float*  SRED = (float*)(smem + SRED_OFF);

    const int tid  = threadIdx.x;
    const int lane = tid & 31;
    const int wv   = tid >> 5;

    // ---- stage all weights into LDS (transposed, bf16) ----
    const float* wsrc[4] = { w1, w2, w3, w4 };
#pragma unroll
    for (int l = 0; l < 4; ++l) {
        __bf16* dst = Wt + l * 16384;
        const float* src = wsrc[l];
        for (int i = tid; i < 16384; i += 256) {
            const int k = i >> 7, n = i & 127;
            dst[n * 128 + k] = f2bf(src[i]);
        }
    }
    for (int i = tid; i < 2048; i += 256)          // cols 0..15 of w5 (only 0,1,2,10 used)
        W5t[i] = f2bf(w5[(i & 127) * OUTDIM + (i >> 7)]);
    if (tid < 128) {
        Bv[tid]       = b1[tid];  Bv[128 + tid] = b2[tid];
        Bv[256 + tid] = b3[tid];  Bv[384 + tid] = b4[tid];
        LNp[tid]       = ln1g[tid]; LNp[128 + tid] = ln1b[tid];
        LNp[256 + tid] = ln2g[tid]; LNp[384 + tid] = ln2b[tid];
    }
    if (tid < 16) { B5s[tid] = b5[tid]; SCs[tid] = scale[tid]; }
    __syncthreads();

    // ---- stage x = instance_feature + anchor_embed for this wave's 16 rows ----
    __bf16* act = (__bf16*)(smem + ACT_OFF) + wv * 2048;
    const int rowBase = blockIdx.x * ROWS_PER_BLOCK + wv * 16;
    for (int i = lane; i < 2048; i += 32) {
        const int r = i >> 7, cidx = i & 127;
        const long g = (long)(rowBase + r) * DDIM + cidx;
        act[i] = f2bf(inst[g] + emb[g]);
    }
    __syncthreads();

    // ---- four hidden layers (in-place per-wave LDS bounce) ----
    mlp_layer(act, act, Wt,          Bv,       lane, false, nullptr, nullptr);
    __syncthreads();
    mlp_layer(act, act, Wt + 16384,  Bv + 128, lane, true,  LNp,        LNp + 128);
    __syncthreads();
    mlp_layer(act, act, Wt + 32768,  Bv + 256, lane, false, nullptr, nullptr);
    __syncthreads();
    mlp_layer(act, act, Wt + 49152,  Bv + 384, lane, true,  LNp + 256,  LNp + 384);
    __syncthreads();

    // ---- output projection: single 16-col tile (cols 0..15 cover 0,1,2,10) ----
    {
        v16bf a[4];
        load_a(act, lane, a);
        const int nlo = lane & 15, h = lane >> 4;
        v8f acc;
        const float bn = B5s[nlo];
#pragma unroll
        for (int r = 0; r < 8; ++r) acc[r] = bn;
#pragma unroll
        for (int kt = 0; kt < 4; ++kt) {
            v16bf b = load_b(W5t, lane, 0, kt);
            acc = __builtin_amdgcn_wmma_f32_16x16x32_bf16(
                false, a[kt], false, b, (short)0, acc, false, false);
        }
        const float sc = SCs[nlo];
        float* ob = (float*)(smem + OUTB_OFF) + wv * 256;
#pragma unroll
        for (int r = 0; r < 8; ++r)
            ob[(r + 8 * h) * 16 + nlo] = acc[r] * sc;
    }
    __syncthreads();

    // ---- per-row postprocessing; only cols 0,1,2,10 feed the outputs ----
    float partial = 0.0f;
    if (lane < 16) {
        const int row = rowBase + lane;
        const float* ob = (const float*)(smem + OUTB_OFF) + wv * 256 + lane * 16;
        const long ab = (long)row * OUTDIM;
        const float o0  = ob[0]  + anchor[ab + 0];
        const float o1  = ob[1]  + anchor[ab + 1];
        const float o2  = ob[2]  + anchor[ab + 2];
        const float o10 = ob[10] + anchor[ab + 10];
        const float x0 = safe_sigmoid(o0);
        const float x1 = safe_sigmoid(o1);
        const float x2 = safe_sigmoid(o2);
        const float op = safe_sigmoid(o10);
        ws_xyz[(long)row * 3 + 0] = x0;
        ws_xyz[(long)row * 3 + 1] = x1;
        ws_xyz[(long)row * 3 + 2] = x2;
        ws_opa[row] = op;
        partial = 1.0f / fabsf(x0) + 1.0f / fabsf(x1) + 1.0f / fabsf(x2);
    }
#pragma unroll
    for (int msk = 1; msk <= 8; msk <<= 1) partial += __shfl_xor(partial, msk);
    if (lane == 0) SRED[wv] = partial;
    __syncthreads();
    if (tid == 0) {
        float s = 0.0f;
#pragma unroll
        for (int i = 0; i < 8; ++i) s += SRED[i];
        ws_part[blockIdx.x] = s;
    }
}

// ---------------------------------------------------------------------------
// Kernel 3: reduce per-block partials -> scalar d = 1/sum(1/|xyz|)
// ---------------------------------------------------------------------------
__global__ void reduce_kernel(const float* __restrict__ part, int n,
                              float* __restrict__ dptr) {
    __shared__ float sm[8];
    float s = 0.0f;
    for (int i = threadIdx.x; i < n; i += 256) s += part[i];
#pragma unroll
    for (int msk = 1; msk < 32; msk <<= 1) s += __shfl_xor(s, msk);
    if ((threadIdx.x & 31) == 0) sm[threadIdx.x >> 5] = s;
    __syncthreads();
    if (threadIdx.x == 0) {
        float t = 0.0f;
#pragma unroll
        for (int i = 0; i < 8; ++i) t += sm[i];
        dptr[0] = 1.0f / t;
    }
}

// ---------------------------------------------------------------------------
// Kernel 4: scatter offsets into means_new / opac_new (idxs are a permutation)
// ---------------------------------------------------------------------------
__global__ void scatter_kernel(const float* __restrict__ ws_xyz,
                               const float* __restrict__ ws_opa,
                               const float* __restrict__ dptr,
                               const float* __restrict__ means,
                               const float* __restrict__ opac,
                               const int* __restrict__ idxs,
                               float* __restrict__ out) {
    const int row = blockIdx.x * blockDim.x + threadIdx.x;
    if (row >= NROWS) return;
    const float d  = dptr[0];
    const float dc = fmaxf(d, 1e-8f);
    const float factor = (expf(d - 6.0f) - expf(-6.0f)) / dc;  // (xyz/dc)*(e^{d-6}-e^{-6})
    const long idx = (long)idxs[row];
#pragma unroll
    for (int c = 0; c < 3; ++c)
        out[idx * 3 + c] = means[idx * 3 + c] + ws_xyz[(long)row * 3 + c] * factor;
    out[GCOUNT * 3 + idx] = opac[idx] + 0.05f * ws_opa[row];
}

// ---------------------------------------------------------------------------
// Host entry
// ---------------------------------------------------------------------------
extern "C" void kernel_launch(void* const* d_in, const int* in_sizes, int n_in,
                              void* d_out, int out_size, void* d_ws, size_t ws_size,
                              hipStream_t stream) {
    const float* inst   = (const float*)d_in[0];
    const float* anchor = (const float*)d_in[1];
    const float* emb    = (const float*)d_in[2];
    const float* means  = (const float*)d_in[3];
    const float* opac   = (const float*)d_in[4];
    const float* w1     = (const float*)d_in[5];
    const float* b1     = (const float*)d_in[6];
    const float* w2     = (const float*)d_in[7];
    const float* b2     = (const float*)d_in[8];
    const float* ln1g   = (const float*)d_in[9];
    const float* ln1b   = (const float*)d_in[10];
    const float* w3     = (const float*)d_in[11];
    const float* b3     = (const float*)d_in[12];
    const float* w4     = (const float*)d_in[13];
    const float* b4     = (const float*)d_in[14];
    const float* ln2g   = (const float*)d_in[15];
    const float* ln2b   = (const float*)d_in[16];
    const float* w5     = (const float*)d_in[17];
    const float* b5     = (const float*)d_in[18];
    const float* scale  = (const float*)d_in[19];
    const int*   idxs   = (const int*)d_in[20];
    float* out = (float*)d_out;

    float* ws      = (float*)d_ws;
    float* ws_xyz  = ws;                       // 3*N floats
    float* ws_opa  = ws + 3L * NROWS;          // N floats
    float* ws_part = ws + 4L * NROWS;          // NBLOCKS floats
    float* ws_d    = ws_part + NBLOCKS;        // 1 float

    // Opt in to >64KB dynamic LDS (gfx1250 WGP has 320KB). Errors ignored.
    (void)hipFuncSetAttribute((const void*)mlp_kernel,
                              hipFuncAttributeMaxDynamicSharedMemorySize,
                              SMEM_TOTAL);

    copy_kernel<<<4096, 256, 0, stream>>>(means, opac, out);
    mlp_kernel<<<NBLOCKS, 256, SMEM_TOTAL, stream>>>(
        inst, anchor, emb, w1, b1, w2, b2, ln1g, ln1b, w3, b3, w4, b4,
        ln2g, ln2b, w5, b5, scale, ws_xyz, ws_opa, ws_part);
    reduce_kernel<<<1, 256, 0, stream>>>(ws_part, NBLOCKS, ws_d);
    scatter_kernel<<<(NROWS + 255) / 256, 256, 0, stream>>>(
        ws_xyz, ws_opa, ws_d, means, opac, idxs, out);
}
